// VanillaRNN_88682484728376
// MI455X (gfx1250) — matrix operand order repaired and checked by reading, not compiled, not run on hardware
//
#include <hip/hip_runtime.h>
#include <hip/hip_bf16.h>
#include <math.h>

// ---------------------------------------------------------------------------
// Vanilla RNN on MI455X (gfx1250):
//   h_{t+1} = tanh(x_t * W_hx + h_t @ W_hh + b_h);  p = h_T @ W_ph + b_p
// B=1024, T=128, H=2048, C=10.
// Strategy: 64 persistent workgroups, each owns 16 batch rows. h tile
// (16x2048 bf16 = 64KB) lives in LDS across all 128 steps. The recurrent
// GEMM uses v_wmma_f32_16x16x32_bf16; W_hh is pre-packed (fp32 -> bf16,
// WMMA-B per-lane tile layout) into d_ws so every B-operand fetch is a
// contiguous 32B-per-lane load served from the 192MB L2.
// This round: epilogue tanh uses the native gfx1250 V_TANH_F32 trans op
// (guarded) instead of the ~20-op OCML expansion, so the per-step epilogue
// co-executes with the XDL pipe instead of stalling it.
// ---------------------------------------------------------------------------

typedef __attribute__((ext_vector_type(16))) __bf16 v16bf;
typedef __attribute__((ext_vector_type(8)))  __bf16 v8bf;
typedef __attribute__((ext_vector_type(8)))  float  v8f;

#if defined(__HIP_DEVICE_COMPILE__) && __has_builtin(__builtin_amdgcn_tanhf)
#define TANHF(v) __builtin_amdgcn_tanhf(v)
#else
#define TANHF(v) tanhf(v)
#endif

constexpr int B_ = 1024;
constexpr int T_ = 128;
constexpr int H_ = 2048;
constexpr int C_ = 10;

constexpr int BM      = 16;              // batch rows per workgroup
constexpr int WAVES   = 8;               // 256 threads = 8 wave32
constexpr int NTILES  = H_ / 16;         // 128 N-tiles of 16 columns
constexpr int NT_PW   = NTILES / WAVES;  // 16 N-tiles per wave
constexpr int KTILES  = H_ / 32;         // 64 K-tiles of 32

// ---------------------------------------------------------------------------
// Pack W_hh (fp32 row-major [H][H], h@W_hh so element (k,j)) into bf16 tiles
// in the WMMA B-matrix (32x16, K x N) per-lane layout:
//   lane = (kk/16)*16 + nn ; element i of the lane's v16bf = B[grp*16+i][nn]
// Each (ktile,ntile) tile occupies 512 contiguous bf16 (1KB); each lane's 16
// values are contiguous -> runtime loads are plain b128 pairs.
// ---------------------------------------------------------------------------
__global__ void pack_whh_kernel(const float* __restrict__ whh,
                                __bf16* __restrict__ packed) {
    int idx = blockIdx.x * 256 + threadIdx.x;
    if (idx >= H_ * H_) return;
    int k = idx / H_;
    int j = idx - k * H_;
    int kt = k >> 5, kk = k & 31;
    int nt = j >> 4, nn = j & 15;
    int lane = ((kk >> 4) << 4) | nn;
    int i    = kk & 15;
    size_t o = ((size_t)(kt * NTILES + nt)) * 512 + (size_t)lane * 16 + i;
    packed[o] = (__bf16)whh[idx];
}

// ---------------------------------------------------------------------------
// Persistent recurrent kernel: one workgroup = 16 batch rows, full H width.
// ---------------------------------------------------------------------------
__global__ void __launch_bounds__(256, 1)
rnn_scan_kernel(const float* __restrict__ x,        // [B,T]
                const float* __restrict__ whx,      // [H]
                const __bf16* __restrict__ whhp,    // packed [H/32][H/16][512]
                const float* __restrict__ bh,       // [H]
                const float* __restrict__ wph,      // [H,C]
                const float* __restrict__ bp,       // [C]
                float* __restrict__ out) {          // [B,C]
    __shared__ __bf16 h_lds[BM * H_];               // 64 KB, row-major [16][2048]

    const int tid  = threadIdx.x;
    const int wave = tid >> 5;
    const int lane = tid & 31;
    const int ln   = lane & 15;      // n within tile / m within A row group
    const int grp  = lane >> 4;      // half-wave group
    const int bBase = blockIdx.x * BM;
    const int nt0   = wave * NT_PW;  // this wave's first N-tile

    // h_0 = 0
    for (int i = tid; i < BM * H_; i += 256) h_lds[i] = (__bf16)0.0f;
    __syncthreads();

    // Hoist per-lane column constants (W_hx, b_h) out of the time loop.
    float whx_r[NT_PW], bh_r[NT_PW];
#pragma unroll
    for (int t = 0; t < NT_PW; ++t) {
        int col = (nt0 + t) * 16 + ln;
        whx_r[t] = whx[col];
        bh_r[t]  = bh[col];
    }

    for (int step = 0; step < T_; ++step) {
        v8f acc[NT_PW] = {};

        // GEMM: [16 x 2048] += h_tile[16 x 2048] @ W_hh[2048 x 2048]
        for (int kt = 0; kt < KTILES; ++kt) {
            // A tile 16x32 bf16 from LDS (contiguous k order, not AMD's table order):
            // lane m=ln, K chunk [16*grp, 16*grp+16) (matches packed B)
            const __bf16* arow = &h_lds[ln * H_ + kt * 32 + grp * 16];
            v8bf alo = *(const v8bf*)(arow);
            v8bf ahi = *(const v8bf*)(arow + 8);
            v16bf a  = __builtin_shufflevector(alo, ahi,
                        0, 1, 2, 3, 4, 5, 6, 7, 8, 9, 10, 11, 12, 13, 14, 15);

            const __bf16* bbase =
                whhp + ((size_t)kt * NTILES + nt0) * 512 + (size_t)lane * 16;
#pragma unroll
            for (int t = 0; t < NT_PW; ++t) {
                v16bf b = *(const v16bf*)(bbase + (size_t)t * 512);
                acc[t] = __builtin_amdgcn_wmma_f32_16x16x32_bf16(
                    /*neg_a=*/false, a, /*neg_b=*/false, b,
                    /*c_mod=*/(short)0, acc[t],
                    /*reuse_a=*/false, /*reuse_b=*/false);
            }
        }

        // x_t values for this lane's 8 output rows (m = r + 8*grp)
        float xv[8];
#pragma unroll
        for (int r = 0; r < 8; ++r)
            xv[r] = x[(size_t)(bBase + r + 8 * grp) * T_ + step];

        __syncthreads();   // all reads of h_t done before overwriting

        // Epilogue: rank-1 input term + bias + tanh, write h_{t+1} (bf16).
        // C-layout: acc[t][r] is (m = r + 8*grp, n = tile_col + ln).
#pragma unroll
        for (int t = 0; t < NT_PW; ++t) {
            int cb = (nt0 + t) * 16 + ln;
#pragma unroll
            for (int r = 0; r < 8; ++r) {
                float v = acc[t][r] + xv[r] * whx_r[t] + bh_r[t];
                h_lds[(r + 8 * grp) * H_ + cb] = (__bf16)TANHF(v);
            }
        }
        __syncthreads();   // h_{t+1} visible to all waves
    }

    // p = h_T @ W_ph + b_p   (16 x 10 per workgroup: negligible, scalar)
    for (int idx = tid; idx < BM * C_; idx += 256) {
        int m = idx / C_;
        int c = idx - m * C_;
        float s = bp[c];
        for (int k = 0; k < H_; ++k)
            s += (float)h_lds[m * H_ + k] * wph[(size_t)k * C_ + c];
        out[(size_t)(bBase + m) * C_ + c] = s;
    }
}

// ---------------------------------------------------------------------------
extern "C" void kernel_launch(void* const* d_in, const int* in_sizes, int n_in,
                              void* d_out, int out_size, void* d_ws, size_t ws_size,
                              hipStream_t stream) {
    const float* x   = (const float*)d_in[0];   // [B,T]
    const float* whx = (const float*)d_in[1];   // [1,H]
    const float* whh = (const float*)d_in[2];   // [H,H]
    const float* bh  = (const float*)d_in[3];   // [H]
    const float* wph = (const float*)d_in[4];   // [H,C]
    const float* bp  = (const float*)d_in[5];   // [1,C]
    float* out = (float*)d_out;

    __bf16* packed = (__bf16*)d_ws;             // needs H*H*2 = 8 MB scratch

    pack_whh_kernel<<<(H_ * H_ + 255) / 256, 256, 0, stream>>>(whh, packed);
    rnn_scan_kernel<<<B_ / BM, 256, 0, stream>>>(x, whx, packed, bh, wph, bp, out);
}
